// TransformerBlock_37778532336011
// MI455X (gfx1250) — compile-verified, matrix operand order checked by reading, not run on hardware
//
#include <hip/hip_runtime.h>
#include <hip/hip_bf16.h>

typedef __attribute__((ext_vector_type(16))) __bf16 v16bf;
typedef __attribute__((ext_vector_type(8)))  float  v8f;
typedef unsigned short u16;
typedef unsigned int   u32;
typedef int v4i_vs __attribute__((vector_size(16)));   // matches builtin's V4i param

#define D_MODEL 768
#define D_HEAD  64
#define N_HEADS 12
#define D_MLP   3072
#define SEQ     4096
#define QKV_N   2304   // 3 * 768, cols = proj*768 + h*64 + d

// ---------- helpers ----------

__device__ __forceinline__ u16 f2bf(float f) {
  u32 u = __float_as_uint(f);
  u32 r = (u + 0x7FFFu + ((u >> 16) & 1u)) >> 16;   // round-to-nearest-even
  return (u16)r;
}

union Frag { uint4 u[2]; v16bf v; };

__device__ __forceinline__ v16bf ld_frag(const u16* p) {
  // 16 bf16: K 0..7 in first b128, K 16..23 in second (CDNA5 A/B frag layout)
  Frag f;
  f.u[0] = *(const uint4*)(p);
  f.u[1] = *(const uint4*)(p + 16);
  return f.v;
}

__device__ __forceinline__ v8f wmma_bf16(v16bf a, v16bf b, v8f c) {
  return __builtin_amdgcn_wmma_f32_16x16x32_bf16(false, a, false, b, (short)0, c, false, false);
}

__device__ __forceinline__ void wait_ds0() {
#if __has_builtin(__builtin_amdgcn_s_wait_dscnt)
  __builtin_amdgcn_s_wait_dscnt(0);
#else
  asm volatile("s_wait_dscnt 0" ::: "memory");
#endif
}

__device__ __forceinline__ void wait_async0() {
#if __has_builtin(__builtin_amdgcn_s_wait_asynccnt)
  __builtin_amdgcn_s_wait_asynccnt(0);
#else
  asm volatile("s_wait_asynccnt 0" ::: "memory");
#endif
}

// Async DMA 16B global -> LDS (ASYNCcnt path); sync fallback if builtin absent.
__device__ __forceinline__ void async_ld_b128(const u16* g, u16* l) {
#if __has_builtin(__builtin_amdgcn_global_load_async_to_lds_b128)
  __builtin_amdgcn_global_load_async_to_lds_b128(
      (__attribute__((address_space(1))) v4i_vs*)(void*)g,
      (__attribute__((address_space(3))) v4i_vs*)(void*)l, 0, 0);
#else
  *(uint4*)l = *(const uint4*)g;
#endif
}

// ---------- weight prep (bf16 + pre-transpose so every GEMM B-tile is row-major in K) ----

// W_Q/K/V: [h, m, d] fp32 -> fused transposed bf16 [2304 n][768 k] ; n = p*768 + h*64 + d
__global__ void qkvw_cast_kernel(const float* __restrict__ wq, const float* __restrict__ wk,
                                 const float* __restrict__ wv, u16* __restrict__ dst) {
  int i = blockIdx.x * blockDim.x + threadIdx.x;
  const int n = N_HEADS * D_MODEL * D_HEAD;
  if (i >= n) return;
  int h = i / (D_MODEL * D_HEAD);
  int r = i - h * (D_MODEL * D_HEAD);
  int m = r / D_HEAD;
  int d = r - m * D_HEAD;
  size_t col = (size_t)h * D_HEAD + d;           // within one projection
  dst[(col)            * D_MODEL + m] = f2bf(wq[i]);
  dst[(col + 768)      * D_MODEL + m] = f2bf(wk[i]);
  dst[(col + 1536)     * D_MODEL + m] = f2bf(wv[i]);
}

// src fp32 [K, N] -> dst bf16 [N, K]
__global__ void transpose_cast_kernel(const float* __restrict__ src, u16* __restrict__ dst,
                                      int K, int N) {
  int i = blockIdx.x * blockDim.x + threadIdx.x;
  if (i >= K * N) return;
  int k = i / N, n = i - k * N;
  dst[(size_t)n * K + k] = f2bf(src[i]);
}

__global__ void concat_bias_kernel(const float* __restrict__ bq, const float* __restrict__ bk,
                                   const float* __restrict__ bv, float* __restrict__ out) {
  int i = blockIdx.x * blockDim.x + threadIdx.x;
  if (i >= QKV_N) return;
  out[i] = (i < 768) ? bq[i] : (i < 1536) ? bk[i - 768] : bv[i - 1536];
}

// ---------- layernorm (one row / workgroup, 768 = 256*3) ----------

__global__ __launch_bounds__(256) void ln_kernel(const float* __restrict__ x,
                                                 const float* __restrict__ w,
                                                 const float* __restrict__ b,
                                                 u16* __restrict__ out) {
  __shared__ float s1[256], s2[256];
  int row = blockIdx.x, t = threadIdx.x;
  const float* xr = x + (size_t)row * D_MODEL;
  float v0 = xr[t], v1 = xr[t + 256], v2 = xr[t + 512];
  s1[t] = v0 + v1 + v2;
  s2[t] = v0 * v0 + v1 * v1 + v2 * v2;
  __syncthreads();
  for (int o = 128; o > 0; o >>= 1) {
    if (t < o) { s1[t] += s1[t + o]; s2[t] += s2[t + o]; }
    __syncthreads();
  }
  float mean = s1[0] * (1.0f / D_MODEL);
  float var  = s2[0] * (1.0f / D_MODEL) - mean * mean;
  float rstd = rsqrtf(var + 1e-5f);
  u16* orow = out + (size_t)row * D_MODEL;
  orow[t]       = f2bf((v0 - mean) * rstd * w[t]       + b[t]);
  orow[t + 256] = f2bf((v1 - mean) * rstd * w[t + 256] + b[t + 256]);
  orow[t + 512] = f2bf((v2 - mean) * rstd * w[t + 512] + b[t + 512]);
}

// ---------- WMMA GEMM: C[M,N] = A[M,K](bf16) * Bt[N,K](bf16)^T + bias (+res) (+gelu) -----
// Double-buffered LDS tiles filled with async global->LDS DMA.

#define BM 128
#define BN 128
#define BK 32
#define AST 40  // LDS row stride in elems (80B, 16B aligned, dodges bank conflicts)

// stage a 128-row x 32-col bf16 tile: 512 b128 chunks, 2 per thread, no exec masking
__device__ __forceinline__ void stage_tile(const u16* __restrict__ G, u16* L,
                                           int rowbase, int K, int k0, int tid) {
  int c0 = tid, c1 = tid + 256;
  int r0 = c0 >> 2, q0 = (c0 & 3) * 8;
  int r1 = c1 >> 2, q1 = (c1 & 3) * 8;
  async_ld_b128(&G[(size_t)(rowbase + r0) * K + k0 + q0], &L[r0 * AST + q0]);
  async_ld_b128(&G[(size_t)(rowbase + r1) * K + k0 + q1], &L[r1 * AST + q1]);
}

__global__ __launch_bounds__(256) void gemm_bf16_kernel(
    const u16* __restrict__ A, const u16* __restrict__ Bt,
    const float* __restrict__ bias, const float* __restrict__ residual,
    float* __restrict__ outf, u16* __restrict__ outh,
    int M, int N, int K, int gelu) {
  __shared__ __align__(16) u16 As[2][BM * AST];
  __shared__ __align__(16) u16 Bs[2][BN * AST];   // [n][k]

  int tid = threadIdx.x, lane = tid & 31, wid = tid >> 5;
  int l15 = lane & 15, half = lane >> 4;
  int bm = blockIdx.x * BM, bn = blockIdx.y * BN;
  int wm = (wid & 3) * 32, wn = (wid >> 2) * 64;   // wave tile 32x64

  v8f acc[2][4];
#pragma unroll
  for (int i = 0; i < 2; i++)
#pragma unroll
    for (int j = 0; j < 4; j++) acc[i][j] = (v8f)(0.0f);

  stage_tile(A, As[0], bm, K, 0, tid);
  stage_tile(Bt, Bs[0], bn, K, 0, tid);

  int nk = K / BK;
  for (int t = 0; t < nk; t++) {
    int buf = t & 1;
    wait_async0();
    __syncthreads();   // stage(t) visible; all waves done reading buf^1
    if (t + 1 < nk) {  // overlap DMA of tile t+1 with WMMA on tile t
      stage_tile(A, As[buf ^ 1], bm, K, (t + 1) * BK, tid);
      stage_tile(Bt, Bs[buf ^ 1], bn, K, (t + 1) * BK, tid);
    }
    if (t + 2 < nk) {  // long-range L2 prefetch (global_prefetch_b8)
      __builtin_prefetch(&A[(size_t)(bm + (tid >> 1)) * K + (t + 2) * BK], 0, 1);
      __builtin_prefetch(&Bt[(size_t)(bn + (tid >> 1)) * K + (t + 2) * BK], 0, 1);
    }

    v16bf af[2], bf4[4];
#pragma unroll
    for (int i = 0; i < 2; i++)
      af[i] = ld_frag(&As[buf][(wm + i * 16 + l15) * AST + half * 8]);
#pragma unroll
    for (int j = 0; j < 4; j++)
      bf4[j] = ld_frag(&Bs[buf][(wn + j * 16 + l15) * AST + half * 8]);
#pragma unroll
    for (int i = 0; i < 2; i++)
#pragma unroll
      for (int j = 0; j < 4; j++)
        acc[i][j] = wmma_bf16(af[i], bf4[j], acc[i][j]);
  }

  // epilogue: C layout — VGPR r -> row r + 8*half, lane l15 -> col
#pragma unroll
  for (int i = 0; i < 2; i++)
#pragma unroll
    for (int j = 0; j < 4; j++)
#pragma unroll
      for (int r = 0; r < 8; r++) {
        int row = bm + wm + i * 16 + r + half * 8;
        int col = bn + wn + j * 16 + l15;
        float v = acc[i][j][r];
        if (bias) v += bias[col];
        if (residual) v += residual[(size_t)row * N + col];
        if (gelu) v = 0.5f * v * (1.0f + erff(v * 0.70710678f));
        size_t o = (size_t)row * N + col;
        if (outf) outf[o] = v;
        if (outh) outh[o] = f2bf(v);
      }
}

// ---------- flash attention: Q,K,V from fused qkv bf16 [SEQ,2304]; Z bf16 [SEQ,768] ------

#define KST 72  // LDS stride for 64-wide K/V chunks (144B rows, 16B aligned)

__global__ __launch_bounds__(256) void attn_kernel(const u16* __restrict__ qkv,
                                                   u16* __restrict__ z) {
  __shared__ __align__(16) u16 Ks[64 * KST];         // [kpos][d]
  __shared__ __align__(16) u16 Vt[64 * KST];         // [d][kpos]
  __shared__ __align__(16) u16 Pw[8 * 16 * KST];     // per-wave P patch [q][kpos]

  int h = blockIdx.x, qb = blockIdx.y;
  int tid = threadIdx.x, lane = tid & 31, wid = tid >> 5;
  int l15 = lane & 15, half = lane >> 4;
  int qrow0 = qb * 128 + wid * 16;
  const size_t ld = QKV_N;

  // resident Q fragments (16 q-rows x 64 d, two K=32 windows)
  v16bf qf[2];
  {
    const u16* qp = qkv + (size_t)(qrow0 + l15) * ld + h * D_HEAD;
    qf[0] = ld_frag(qp + half * 8);
    qf[1] = ld_frag(qp + 32 + half * 8);
  }

  float m_i[8], l_i[8];
  v8f zacc[4];
#pragma unroll
  for (int r = 0; r < 8; r++) { m_i[r] = -3.0e38f; l_i[r] = 0.0f; }
#pragma unroll
  for (int dn = 0; dn < 4; dn++) zacc[dn] = (v8f)(0.0f);

  int nkc = 2 * qb + 2;   // causal: 64-wide chunks up to the block diagonal
  for (int kc = 0; kc < nkc; kc++) {
    __syncthreads();   // previous compute done before overwriting Ks/Vt
    // stage K chunk [kpos][d] via async DMA: 512 chunks, 2/thread
    {
      int c0 = tid, c1 = tid + 256;
      int kr0 = c0 >> 3, dq0 = (c0 & 7) * 8;
      int kr1 = c1 >> 3, dq1 = (c1 & 7) * 8;
      async_ld_b128(&qkv[(size_t)(kc * 64 + kr0) * ld + 768 + h * D_HEAD + dq0],
                    &Ks[kr0 * KST + dq0]);
      async_ld_b128(&qkv[(size_t)(kc * 64 + kr1) * ld + 768 + h * D_HEAD + dq1],
                    &Ks[kr1 * KST + dq1]);
    }
    // stage V transposed -> Vt[d][kpos] (needs element scatter; sync path)
#pragma unroll
    for (int c = tid; c < 512; c += 256) {
      int kr = c >> 3, dq = (c & 7) * 8;
      union { uint4 u; u16 hh[8]; } t8;
      t8.u = *(const uint4*)&qkv[(size_t)(kc * 64 + kr) * ld + 1536 + h * D_HEAD + dq];
#pragma unroll
      for (int e = 0; e < 8; e++) Vt[(dq + e) * KST + kr] = t8.hh[e];
    }
    wait_async0();
    __syncthreads();

    // S = Q K^T for this 16q x 64k tile (4 N-frags, contraction d=64 = 2 WMMAs)
    v8f s[4];
#pragma unroll
    for (int kn = 0; kn < 4; kn++) {
      s[kn] = (v8f)(0.0f);
      v16bf kb0 = ld_frag(&Ks[(kn * 16 + l15) * KST + half * 8]);
      v16bf kb1 = ld_frag(&Ks[(kn * 16 + l15) * KST + 32 + half * 8]);
      s[kn] = wmma_bf16(qf[0], kb0, s[kn]);
      s[kn] = wmma_bf16(qf[1], kb1, s[kn]);
    }

    // scale, causal mask, online softmax (row = r + 8*half, col from lane)
    float rscale[8];
#pragma unroll
    for (int r = 0; r < 8; r++) {
      int rowq = qrow0 + r + half * 8;
      float mx = -3.0e38f;
#pragma unroll
      for (int kn = 0; kn < 4; kn++) {
        int col = kc * 64 + kn * 16 + l15;
        float e = s[kn][r] * 0.125f;            // 1/sqrt(64)
        if (col > rowq) e = -100000.0f;         // IGNORE
        s[kn][r] = e;
        mx = fmaxf(mx, e);
      }
#pragma unroll
      for (int o = 1; o < 16; o <<= 1) mx = fmaxf(mx, __shfl_xor(mx, o, 32));
      float mnew = fmaxf(m_i[r], mx);
      float sc = __expf(m_i[r] - mnew);
      float rsum = 0.0f;
#pragma unroll
      for (int kn = 0; kn < 4; kn++) {
        float p = __expf(s[kn][r] - mnew);
        s[kn][r] = p;
        rsum += p;
      }
#pragma unroll
      for (int o = 1; o < 16; o <<= 1) rsum += __shfl_xor(rsum, o, 32);
      l_i[r] = l_i[r] * sc + rsum;
      m_i[r] = mnew;
      rscale[r] = sc;
    }
#pragma unroll
    for (int dn = 0; dn < 4; dn++)
#pragma unroll
      for (int r = 0; r < 8; r++) zacc[dn][r] *= rscale[r];

    // P -> per-wave LDS (bf16), then re-fragment as A-operand
    u16* pw = &Pw[wid * 16 * KST];
#pragma unroll
    for (int kn = 0; kn < 4; kn++)
#pragma unroll
      for (int r = 0; r < 8; r++)
        pw[(r + half * 8) * KST + kn * 16 + l15] = f2bf(s[kn][r]);
    wait_ds0();

    v16bf pa[2];
    pa[0] = ld_frag(&pw[l15 * KST + half * 8]);
    pa[1] = ld_frag(&pw[l15 * KST + 32 + half * 8]);
#pragma unroll
    for (int dn = 0; dn < 4; dn++) {
      v16bf vb0 = ld_frag(&Vt[(dn * 16 + l15) * KST + half * 8]);
      v16bf vb1 = ld_frag(&Vt[(dn * 16 + l15) * KST + 32 + half * 8]);
      zacc[dn] = wmma_bf16(pa[0], vb0, zacc[dn]);
      zacc[dn] = wmma_bf16(pa[1], vb1, zacc[dn]);
    }
  }

  // normalize and store Z (bf16) at [row, h*64 + d]
#pragma unroll
  for (int dn = 0; dn < 4; dn++)
#pragma unroll
    for (int r = 0; r < 8; r++) {
      int row = qrow0 + r + half * 8;
      int col = h * D_HEAD + dn * 16 + l15;
      z[(size_t)row * D_MODEL + col] = f2bf(zacc[dn][r] / l_i[r]);
    }
}

// ---------- launch ----------

extern "C" void kernel_launch(void* const* d_in, const int* in_sizes, int n_in,
                              void* d_out, int out_size, void* d_ws, size_t ws_size,
                              hipStream_t stream) {
  const float* resid_pre = (const float*)d_in[0];
  const float* ln1_w = (const float*)d_in[1];
  const float* ln1_b = (const float*)d_in[2];
  const float* ln2_w = (const float*)d_in[3];
  const float* ln2_b = (const float*)d_in[4];
  const float* W_Q  = (const float*)d_in[5];
  const float* W_K  = (const float*)d_in[6];
  const float* W_V  = (const float*)d_in[7];
  const float* W_O  = (const float*)d_in[8];
  const float* b_Q  = (const float*)d_in[9];
  const float* b_K  = (const float*)d_in[10];
  const float* b_V  = (const float*)d_in[11];
  const float* b_O  = (const float*)d_in[12];
  const float* W_in  = (const float*)d_in[13];
  const float* b_in  = (const float*)d_in[14];
  const float* W_out = (const float*)d_in[15];
  const float* b_out = (const float*)d_in[16];

  float* out_resid = (float*)d_out;                          // [4096, 768]
  float* out_post  = (float*)d_out + (size_t)SEQ * D_MODEL;  // [4096, 3072]

  char* base = (char*)d_ws;
  size_t off = 0;
  auto alloc = [&](size_t bytes) -> void* {
    void* p = base + off;
    off = (off + bytes + 255) & ~(size_t)255;
    return p;
  };
  u16* wqkvT = (u16*)alloc((size_t)QKV_N * D_MODEL * 2);    // [2304][768]
  u16* woT   = (u16*)alloc((size_t)D_MODEL * D_MODEL * 2);  // [768][768]
  u16* winT  = (u16*)alloc((size_t)D_MLP * D_MODEL * 2);    // [3072][768]
  u16* woutT = (u16*)alloc((size_t)D_MODEL * D_MLP * 2);    // [768][3072]
  float* bqkv = (float*)alloc((size_t)QKV_N * 4);
  u16* xln1  = (u16*)alloc((size_t)SEQ * D_MODEL * 2);
  u16* qkv   = (u16*)alloc((size_t)SEQ * QKV_N * 2);
  u16* zb    = (u16*)alloc((size_t)SEQ * D_MODEL * 2);
  float* resid_mid = (float*)alloc((size_t)SEQ * D_MODEL * 4);
  u16* xln2  = (u16*)alloc((size_t)SEQ * D_MODEL * 2);
  u16* postb = (u16*)alloc((size_t)SEQ * D_MLP * 2);

  // weight prep (bf16 + transpose)
  {
    int n = N_HEADS * D_MODEL * D_HEAD;  // 589824
    qkvw_cast_kernel<<<(n + 255) / 256, 256, 0, stream>>>(W_Q, W_K, W_V, wqkvT);
    transpose_cast_kernel<<<(D_MODEL * D_MODEL + 255) / 256, 256, 0, stream>>>(
        W_O, woT, D_MODEL, D_MODEL);
    transpose_cast_kernel<<<(D_MODEL * D_MLP + 255) / 256, 256, 0, stream>>>(
        W_in, winT, D_MODEL, D_MLP);
    transpose_cast_kernel<<<(D_MLP * D_MODEL + 255) / 256, 256, 0, stream>>>(
        W_out, woutT, D_MLP, D_MODEL);
    concat_bias_kernel<<<(QKV_N + 255) / 256, 256, 0, stream>>>(b_Q, b_K, b_V, bqkv);
  }

  // LN1 -> bf16
  ln_kernel<<<SEQ, 256, 0, stream>>>(resid_pre, ln1_w, ln1_b, xln1);

  // QKV = LN1 @ Wqkv + bqkv  (bf16 out only)
  gemm_bf16_kernel<<<dim3(SEQ / BM, QKV_N / BN), 256, 0, stream>>>(
      xln1, wqkvT, bqkv, nullptr, nullptr, qkv, SEQ, QKV_N, D_MODEL, 0);

  // flash attention -> Z bf16
  attn_kernel<<<dim3(N_HEADS, SEQ / 128), 256, 0, stream>>>(qkv, zb);

  // resid_mid = Z @ W_O + b_O + resid_pre  (fp32)
  gemm_bf16_kernel<<<dim3(SEQ / BM, D_MODEL / BN), 256, 0, stream>>>(
      zb, woT, b_O, resid_pre, resid_mid, nullptr, SEQ, D_MODEL, D_MODEL, 0);

  // LN2 -> bf16
  ln_kernel<<<SEQ, 256, 0, stream>>>(resid_mid, ln2_w, ln2_b, xln2);

  // post = gelu(LN2 @ W_in + b_in): fp32 to d_out (output #2) + bf16 mirror
  gemm_bf16_kernel<<<dim3(SEQ / BM, D_MLP / BN), 256, 0, stream>>>(
      xln2, winT, b_in, nullptr, out_post, postb, SEQ, D_MLP, D_MODEL, 1);

  // resid_post = post @ W_out + b_out + resid_mid  (output #1)
  gemm_bf16_kernel<<<dim3(SEQ / BM, D_MODEL / BN), 256, 0, stream>>>(
      postb, woutT, b_out, resid_mid, out_resid, nullptr, SEQ, D_MODEL, D_MLP, 0);

  (void)in_sizes; (void)n_in; (void)out_size; (void)ws_size;
}
